// Conv2DManual_18030272708851
// MI455X (gfx1250) — matrix operand order repaired and checked by reading, not compiled, not run on hardware
//
#include <hip/hip_runtime.h>

typedef __attribute__((ext_vector_type(16))) __bf16 v16bf;
typedef __attribute__((ext_vector_type(8)))  float  v8f;
typedef __attribute__((ext_vector_type(8)))  int    v8i;

#define HH   56
#define WW   56
#define HWP  3136
#define CIN  128
#define COUT 256
#define NRS  9

#define BM   128            // out-channel tile per block
#define BN   128            // spatial tile per block
#define BK   32             // WMMA K step (channels)
#define LDA  40             // padded LDS row stride (halves): 80 B, keeps b128 16B-aligned
#define LDB  40
#define NSTEP (NRS * (CIN / BK))   // 36

// Truncating f32->bf16 pack of two values via one v_perm_b32.
__device__ __forceinline__ unsigned pk2(float a, float b) {
  return __builtin_amdgcn_perm(__builtin_bit_cast(unsigned, b),
                               __builtin_bit_cast(unsigned, a), 0x07060302u);
}

// CDNA5 async memory->LDS copy (ASYNCcnt-tracked), bypasses VGPRs.
__device__ __forceinline__ void async_b128(unsigned lds_off, const __bf16* g) {
  asm volatile("global_load_async_to_lds_b128 %0, %1, off"
               :: "v"(lds_off), "v"(g) : "memory");
}
__device__ __forceinline__ void wait_async0() {
  asm volatile("s_wait_asynccnt 0x0" ::: "memory");
}

// ---- one-time weight transpose/convert: OIHW f32 -> [rs][ko][c] bf16 pairs ----
__global__ __launch_bounds__(256) void cvt_wgt(const float* __restrict__ wgt,
                                               unsigned* __restrict__ wsA) {
  const int i   = blockIdx.x * 256 + threadIdx.x;     // pair index, 9*256*64 total
  const int rs  = i / (COUT * CIN / 2);
  const int rem = i % (COUT * CIN / 2);
  const int ko  = rem / (CIN / 2);
  const int c2  = (rem % (CIN / 2)) * 2;
  const float* wp = wgt + (size_t)ko * (CIN * NRS) + rs;
  wsA[i] = pk2(wp[(size_t)c2 * NRS], wp[(size_t)(c2 + 1) * NRS]);
}

template <bool USE_WS>
__global__ __launch_bounds__(256) void conv3x3_wmma(
    const float* __restrict__ x, const float* __restrict__ wgt,
    const __bf16* __restrict__ wsA, const float* __restrict__ bias,
    float* __restrict__ out) {
  __shared__ __bf16 lA[2][BM * LDA];   // [buf][ko][c] bf16
  __shared__ __bf16 lB[2][BN * LDB];   // [buf][n][c]  bf16

  const int t    = threadIdx.x;
  const int lane = t & 31;
  const int wave = t >> 5;
  const int waveM = wave & 3;          // 4 M-tiles of 32
  const int waveN = wave >> 2;         // 2 N-tiles of 64

  const int pBase  = blockIdx.x * BN;  // spatial base (GEMM N)
  const int koBase = blockIdx.y * BM;  // out-channel base (GEMM M)

  // A staging: 2 threads per ko row, 16 halves each
  const int ako = t >> 1;              // 0..127
  const int ach = (t & 1) * 16;        // 0 or 16
  // B staging: 128 spatial x 32 ch, one spatial point x 16 ch per thread
  const int nl = t & 127;
  const int cb = (t >> 7) * 16;        // 0 or 16
  const int p  = pBase + nl;
  const int pw = p % WW;
  const int ph = (p / WW) % HH;
  const int pn = p / HWP;
  const float* __restrict__ xpn = x + (size_t)pn * CIN * HWP;

  // LDS byte offsets of this thread's A-staging slots (generic ptr low 32 bits
  // are the LDS offset per ISA 10.2 aperture mapping).
  const unsigned ldsA[2] = {
      (unsigned)(size_t)&lA[0][ako * LDA + ach],
      (unsigned)(size_t)&lA[1][ako * LDA + ach]};

  v8f acc[2][4] = {};                  // [msub][nsub]

  // prefetch registers (A regs only used by the non-workspace fallback)
  int4  rA0, rA1;
  float fX[16];

  auto stageA = [&](int step, int buf) {
    const int rs = step >> 2;
    const int ck = (step & 3) * BK;
    if constexpr (USE_WS) {
      const __bf16* wa = wsA + (size_t)rs * (COUT * CIN)
                             + (size_t)(koBase + ako) * CIN + ck + ach;
      async_b128(ldsA[buf], wa);           // 16 B memory -> LDS, no VGPRs
      async_b128(ldsA[buf] + 16, wa + 8);
    } else {
      const float* wb = wgt + (size_t)(koBase + ako) * (CIN * NRS) + rs;
      float f[16];
#pragma unroll
      for (int i = 0; i < 16; ++i) f[i] = wb[(size_t)(ck + ach + i) * NRS];
      rA0.x = pk2(f[0], f[1]);   rA0.y = pk2(f[2], f[3]);
      rA0.z = pk2(f[4], f[5]);   rA0.w = pk2(f[6], f[7]);
      rA1.x = pk2(f[8], f[9]);   rA1.y = pk2(f[10], f[11]);
      rA1.z = pk2(f[12], f[13]); rA1.w = pk2(f[14], f[15]);
    }
  };

  auto prefetchB = [&](int step) {
    const int rs = step >> 2;
    const int ck = (step & 3) * BK;
    const int ih = ph + rs / 3 - 1;
    const int iw = pw + rs % 3 - 1;
    const bool valid = ((unsigned)ih < (unsigned)HH) & ((unsigned)iw < (unsigned)WW);
    const float* __restrict__ xb = xpn + (long)ih * WW + iw;
#pragma unroll
    for (int i = 0; i < 16; ++i)
      fX[i] = valid ? xb[(size_t)(ck + cb + i) * HWP] : 0.0f;
  };

  auto commit = [&](int buf) {
    if constexpr (!USE_WS) {
      *(int4*)&lA[buf][ako * LDA + ach]     = rA0;
      *(int4*)&lA[buf][ako * LDA + ach + 8] = rA1;
    }
    int4 v0, v1;
    v0.x = pk2(fX[0],  fX[1]);  v0.y = pk2(fX[2],  fX[3]);
    v0.z = pk2(fX[4],  fX[5]);  v0.w = pk2(fX[6],  fX[7]);
    v1.x = pk2(fX[8],  fX[9]);  v1.y = pk2(fX[10], fX[11]);
    v1.z = pk2(fX[12], fX[13]); v1.w = pk2(fX[14], fX[15]);
    *(int4*)&lB[buf][nl * LDB + cb]     = v0;
    *(int4*)&lB[buf][nl * LDB + cb + 8] = v1;
  };

  auto ldfrag = [](const __bf16* ptr, int hiOff) -> v16bf {
    const int4 lo = *(const int4*)ptr;
    const int4 hi = *(const int4*)(ptr + hiOff);
    v8i u;
    u[0] = lo.x; u[1] = lo.y; u[2] = lo.z; u[3] = lo.w;
    u[4] = hi.x; u[5] = hi.y; u[6] = hi.z; u[7] = hi.w;
    return __builtin_bit_cast(v16bf, u);
  };

  // fragment LDS offsets (halves)
  // A 16x32 bf16 layout: lane<16 -> K0-7 & 16-23 ; lane>=16 -> K8-15 & 24-31
  const int mrow0 = (waveM * 32 + (lane & 15)) * LDA + (lane >> 4) * 8;
  const int mrow1 = mrow0 + 16 * LDA;
  // B 32x16 bf16 layout: lane = column n; lane<16 -> K0-7 & 16-23 ; lane>=16 -> K8-15 & 24-31
  const int nrow0 = (waveN * 64 + (lane & 15)) * LDB + (lane >> 4) * 8;

  // ---- software pipeline: one barrier per step, async A + reg-staged B ----
  stageA(0, 0);
  prefetchB(0);
  commit(0);

  for (int step = 0; step < NSTEP; ++step) {
    const int buf = step & 1;
    if constexpr (USE_WS) wait_async0();   // our async writes into `buf` landed
    __syncthreads();                       // publish buf; buf^1 free for overwrite
    if (step + 1 < NSTEP) {
      stageA(step + 1, buf ^ 1);           // async: straight into LDS buf^1
      prefetchB(step + 1);                 // f32 -> regs (packed at commit)
    }

    const v16bf a0 = ldfrag(&lA[buf][mrow0], 16);
    const v16bf a1 = ldfrag(&lA[buf][mrow1], 16);
    v16bf b[4];
#pragma unroll
    for (int ns = 0; ns < 4; ++ns)
      b[ns] = ldfrag(&lB[buf][nrow0 + ns * 16 * LDB], 16);

#pragma unroll
    for (int ns = 0; ns < 4; ++ns) {
      acc[0][ns] = __builtin_amdgcn_wmma_f32_16x16x32_bf16(false, a0, false, b[ns],
                                                           (short)0, acc[0][ns], false, false);
      acc[1][ns] = __builtin_amdgcn_wmma_f32_16x16x32_bf16(false, a1, false, b[ns],
                                                           (short)0, acc[1][ns], false, false);
    }

    if (step + 1 < NSTEP) commit(buf ^ 1);
  }

  // ---- epilogue: bias + store 8 x (16x16) f32 tiles, NCHW ----
  // C/D layout: lane<16 -> N=lane, M=v ; lane>=16 -> N=lane-16, M=v+8
#pragma unroll
  for (int ns = 0; ns < 4; ++ns) {
    const int pOut = pBase + waveN * 64 + ns * 16 + (lane & 15);
    const int on   = pOut / HWP;
    const int ohw  = pOut % HWP;
#pragma unroll
    for (int ms = 0; ms < 2; ++ms) {
      const int ko0 = koBase + waveM * 32 + ms * 16 + (lane >> 4) * 8;
      float* __restrict__ ob = out + ((size_t)on * COUT + ko0) * HWP + ohw;
#pragma unroll
      for (int v = 0; v < 8; ++v)
        ob[(size_t)v * HWP] = acc[ms][ns][v] + bias[ko0 + v];
    }
  }
}

extern "C" void kernel_launch(void* const* d_in, const int* in_sizes, int n_in,
                              void* d_out, int out_size, void* d_ws, size_t ws_size,
                              hipStream_t stream) {
  (void)in_sizes; (void)n_in; (void)out_size;
  const float* x    = (const float*)d_in[0];   // (32,128,56,56)
  const float* wgt  = (const float*)d_in[1];   // (256,128,3,3)
  const float* bias = (const float*)d_in[2];   // (256,)
  float* out = (float*)d_out;                  // (32,256,56,56)

  const int nSpatial = 32 * HWP;               // 100352
  dim3 grid(nSpatial / BN, COUT / BM, 1);      // (784, 2)
  dim3 block(256, 1, 1);

  const size_t wsNeed = (size_t)NRS * COUT * CIN * 2;   // 576 KB of bf16 weights
  if (ws_size >= wsNeed) {
    cvt_wgt<<<(NRS * COUT * CIN / 2) / 256, 256, 0, stream>>>(wgt, (unsigned*)d_ws);
    conv3x3_wmma<true><<<grid, block, 0, stream>>>(x, wgt, (const __bf16*)d_ws, bias, out);
  } else {
    conv3x3_wmma<false><<<grid, block, 0, stream>>>(x, wgt, nullptr, bias, out);
  }
}
